// SDPACustom_5308579578003
// MI455X (gfx1250) — matrix operand order repaired and checked by reading, not compiled, not run on hardware
//
#include <hip/hip_runtime.h>

// ---------------- CDNA5 WMMA types ----------------
typedef __attribute__((ext_vector_type(16))) __bf16 v16bf;
typedef __attribute__((ext_vector_type(8)))  float  v8f;

#define B_  4
#define S_  128
#define H_  32
#define D_  128
#define T_  2048
#define KT  64          // keys per KV tile
#define NTHREADS 256    // 8 wave32

// LDS row strides (ushort units); row-stride*2 must be a multiple of 16 bytes
#define QSTR 136        // 272 B
#define KSTR 144        // 288 B (d=128 + pad)
#define VSTR 72         // 144 B (key=64 + pad)
#define PSTR 72         // 144 B

struct Stage2 {
  unsigned short k [KT][KSTR];   // [key][d]      bf16
  unsigned short vt[D_][VSTR];   // [d][key]      bf16 (transposed V)
  unsigned short p [S_][PSTR];   // [q_row][key]  bf16 (per-wave P scratch)
};
union SharedMem {
  unsigned short q[S_][QSTR];    // stage 1 only (dead after A-frag preload)
  Stage2 s2;
};

union Frag16 { uint4 u[2]; v16bf v; };

// f32 -> bf16 via compiler lowering (hardware v_cvt_pk_bf16_f32 class op if present)
__device__ __forceinline__ unsigned short f2bf(float x) {
  return __builtin_bit_cast(unsigned short, (__bf16)x);
}
__device__ __forceinline__ unsigned int f2bf_pack(float lo, float hi) {
  return (unsigned int)f2bf(lo) | ((unsigned int)f2bf(hi) << 16);
}

// A-matrix fragment: two 8-element runs (16B ds_load_b128 each)
__device__ __forceinline__ v16bf ld_frag2(const unsigned short* base, int off0, int off1) {
  Frag16 f;
  f.u[0] = *(const uint4*)(base + off0);
  f.u[1] = *(const uint4*)(base + off1);
  return f.v;
}
// B-matrix fragment: 16 contiguous elements (32B)
__device__ __forceinline__ v16bf ld_frag(const unsigned short* base, int off) {
  return ld_frag2(base, off, off + 8);
}

__device__ __forceinline__ v8f zero8() {
  v8f z;
  #pragma unroll
  for (int i = 0; i < 8; ++i) z[i] = 0.0f;
  return z;
}

__global__ __launch_bounds__(NTHREADS, 1)
void fa_sdpa_kvcache_kernel(const float* __restrict__ qg,
                            const float* __restrict__ kg,
                            const float* __restrict__ vg,
                            const float* __restrict__ kc,
                            const float* __restrict__ vc,
                            const int*   __restrict__ input_pos_p,
                            float*       __restrict__ out)
{
  __shared__ SharedMem sm;

  const int tid  = threadIdx.x;
  const int lane = tid & 31;
  const int wave = tid >> 5;        // 0..7, owns query rows [16w,16w+16)
  const int half = lane >> 4;       // lane-half (A/B K split, C/D M split)
  const int l16  = lane & 15;

  const int bh = blockIdx.x;
  const int b  = bh / H_;
  const int h  = bh % H_;

  const int input_pos = input_pos_p[0];
  const int nTiles    = (input_pos + S_ + KT - 1) / KT;
  const float qscale  = 0.08838834764831845f;   // 1/sqrt(D)

  // ---------- Stage 1: Q tile (128x128 f32 -> bf16, pre-scaled) ----------
  {
    const int row = tid >> 1;                 // 0..127
    const int cb  = (tid & 1) * 64;
    const float* src = qg + (((size_t)b * S_ + row) * H_ + h) * D_;
    #pragma unroll
    for (int it = 0; it < 16; ++it) {
      const int c = cb + it * 4;
      float4 f = *(const float4*)(src + c);
      uint2 o;
      o.x = f2bf_pack(f.x * qscale, f.y * qscale);
      o.y = f2bf_pack(f.z * qscale, f.w * qscale);
      *(uint2*)&sm.q[row][c] = o;
    }
  }
  __syncthreads();

  // Preload Q A-fragments (16 rows x 128 K per wave -> 4 frags, register-resident)
  v16bf aq[4];
  {
    const unsigned short* qrow = sm.q[wave * 16 + l16];
    #pragma unroll
    for (int kk = 0; kk < 4; ++kk)
      aq[kk] = ld_frag2(qrow, kk * 32 + half * 8, kk * 32 + 16 + half * 8);
  }
  __syncthreads();   // sm.q dead; its space is reused by Stage2 buffers

  // Flash-attention state (C/D layout: element vv <-> query row 16w + 8*half + vv)
  v8f acc[8];
  #pragma unroll
  for (int i = 0; i < 8; ++i) acc[i] = zero8();
  float mrow[8], lrow[8];
  #pragma unroll
  for (int i = 0; i < 8; ++i) { mrow[i] = -1.0e30f; lrow[i] = 0.0f; }

  for (int t = 0; t < nTiles; ++t) {
    const int g0 = t * KT;

    // ---------- load K tile (64x128) -> sm.s2.k, bf16 ----------
    {
      const int row = tid >> 2;               // 0..63
      const int cb  = (tid & 3) * 32;
      const int g   = g0 + row;
      int rn = g - input_pos; rn = rn < 0 ? 0 : (rn >= S_ ? S_ - 1 : rn);
      const float* src = (g < input_pos)
          ? kc + (((size_t)b * T_ + g ) * H_ + h) * D_
          : kg + (((size_t)b * S_ + rn) * H_ + h) * D_;
      #pragma unroll
      for (int it = 0; it < 8; ++it) {
        const int c = cb + it * 4;
        float4 f = *(const float4*)(src + c);
        uint2 o;
        o.x = f2bf_pack(f.x, f.y);
        o.y = f2bf_pack(f.z, f.w);
        *(uint2*)&sm.s2.k[row][c] = o;
      }
      if (t + 1 < nTiles) {                   // CDNA5 global_prefetch_b8
        const int g2 = g + KT;
        int rn2 = g2 - input_pos; rn2 = rn2 < 0 ? 0 : (rn2 >= S_ ? S_ - 1 : rn2);
        const float* nsrc = (g2 < input_pos)
            ? kc + (((size_t)b * T_ + g2 ) * H_ + h) * D_
            : kg + (((size_t)b * S_ + rn2) * H_ + h) * D_;
        __builtin_prefetch(nsrc + cb, 0, 1);
      }
    }
    // ---------- load V tile (64x128) transposed -> sm.s2.vt[d][key] ----------
    // Each thread handles two adjacent keys so transposed writes are packed b32.
    {
      const int kp  = tid >> 3;               // 0..31 -> keys 2kp, 2kp+1
      const int cb  = (tid & 7) * 16;         // d base
      const int ga  = g0 + 2 * kp;
      const int gbq = ga + 1;
      int ra = ga  - input_pos; ra = ra < 0 ? 0 : (ra >= S_ ? S_ - 1 : ra);
      int rb = gbq - input_pos; rb = rb < 0 ? 0 : (rb >= S_ ? S_ - 1 : rb);
      const float* s0 = (ga < input_pos)
          ? vc + (((size_t)b * T_ + ga) * H_ + h) * D_
          : vg + (((size_t)b * S_ + ra) * H_ + h) * D_;
      const float* s1 = (gbq < input_pos)
          ? vc + (((size_t)b * T_ + gbq) * H_ + h) * D_
          : vg + (((size_t)b * S_ + rb ) * H_ + h) * D_;
      #pragma unroll
      for (int it = 0; it < 4; ++it) {
        const int c = cb + it * 4;
        float4 f0 = *(const float4*)(s0 + c);
        float4 f1 = *(const float4*)(s1 + c);
        *(unsigned int*)&sm.s2.vt[c + 0][2 * kp] = f2bf_pack(f0.x, f1.x);
        *(unsigned int*)&sm.s2.vt[c + 1][2 * kp] = f2bf_pack(f0.y, f1.y);
        *(unsigned int*)&sm.s2.vt[c + 2][2 * kp] = f2bf_pack(f0.z, f1.z);
        *(unsigned int*)&sm.s2.vt[c + 3][2 * kp] = f2bf_pack(f0.w, f1.w);
      }
      if (t + 1 < nTiles) {
        const int g2 = ga + KT;
        int rn2 = g2 - input_pos; rn2 = rn2 < 0 ? 0 : (rn2 >= S_ ? S_ - 1 : rn2);
        const float* nsrc = (g2 < input_pos)
            ? vc + (((size_t)b * T_ + g2 ) * H_ + h) * D_
            : vg + (((size_t)b * S_ + rn2) * H_ + h) * D_;
        __builtin_prefetch(nsrc + cb, 0, 1);
      }
    }
    __syncthreads();

    // ---------- S = Q @ K^T : 4 n-blocks x 4 k-steps = 16 WMMA ----------
    v8f sc[4];
    #pragma unroll
    for (int nb = 0; nb < 4; ++nb) {
      v8f c = zero8();
      #pragma unroll
      for (int kk = 0; kk < 4; ++kk) {
        v16bf bk = ld_frag2(sm.s2.k[nb * 16 + l16], kk * 32 + half * 8, kk * 32 + 16 + half * 8);
        c = __builtin_amdgcn_wmma_f32_16x16x32_bf16(
                false, aq[kk], false, bk, (short)0, c, false, false);
      }
      sc[nb] = c;
    }

    // ---------- causal mask (only tiles reaching past input_pos) ----------
    if (g0 + KT - 1 > input_pos) {            // wave-uniform branch
      #pragma unroll
      for (int nb = 0; nb < 4; ++nb) {
        const int gkey = g0 + nb * 16 + l16;
        #pragma unroll
        for (int vv = 0; vv < 8; ++vv) {
          const int srow = wave * 16 + half * 8 + vv;     // query local row
          sc[nb][vv] = (gkey > input_pos + srow) ? -3.0e38f : sc[nb][vv];
        }
      }
    }

    // ---------- online softmax (per-row, rows replicated per lane-half) ----------
    float pr[4][8];
    #pragma unroll
    for (int vv = 0; vv < 8; ++vv) {
      float rm = fmaxf(fmaxf(sc[0][vv], sc[1][vv]), fmaxf(sc[2][vv], sc[3][vv]));
      #pragma unroll
      for (int off = 1; off <= 8; off <<= 1)
        rm = fmaxf(rm, __shfl_xor(rm, off, 32));
      const float mnew = fmaxf(mrow[vv], rm);
      const float corr = __expf(mrow[vv] - mnew);
      mrow[vv] = mnew;
      float rs = 0.0f;
      #pragma unroll
      for (int nb = 0; nb < 4; ++nb) {
        const float p = __expf(sc[nb][vv] - mnew);
        pr[nb][vv] = p;
        rs += p;
      }
      #pragma unroll
      for (int off = 1; off <= 8; off <<= 1)
        rs += __shfl_xor(rs, off, 32);
      lrow[vv] = lrow[vv] * corr + rs;
      #pragma unroll
      for (int nb = 0; nb < 8; ++nb)
        acc[nb][vv] *= corr;
    }

    // ---------- P: C-layout -> A-layout via per-wave LDS scratch ----------
    #pragma unroll
    for (int nb = 0; nb < 4; ++nb)
      #pragma unroll
      for (int vv = 0; vv < 8; ++vv)
        sm.s2.p[wave * 16 + half * 8 + vv][nb * 16 + l16] = f2bf(pr[nb][vv]);
    // same-wave LDS ops are ordered; compiler inserts s_wait_dscnt for the RAW

    // ---------- O += P @ V : 2 k-steps x 8 n-blocks = 16 WMMA ----------
    #pragma unroll
    for (int kk = 0; kk < 2; ++kk) {
      v16bf ap = ld_frag2(sm.s2.p[wave * 16 + l16],
                          kk * 32 + half * 8, kk * 32 + 16 + half * 8);
      #pragma unroll
      for (int nb = 0; nb < 8; ++nb) {
        v16bf bv = ld_frag2(sm.s2.vt[nb * 16 + l16], kk * 32 + half * 8, kk * 32 + 16 + half * 8);
        acc[nb] = __builtin_amdgcn_wmma_f32_16x16x32_bf16(
                      false, ap, false, bv, (short)0, acc[nb], false, false);
      }
    }
    __syncthreads();   // protect sm.s2.k / sm.s2.vt before next tile's loads
  }

  // ---------- epilogue: out[b, s, h*D + d] = acc / l ----------
  #pragma unroll
  for (int vv = 0; vv < 8; ++vv) {
    const float inv = 1.0f / lrow[vv];
    const int srow  = wave * 16 + half * 8 + vv;
    float* orow = out + ((size_t)b * S_ + srow) * (H_ * D_) + (size_t)h * D_;
    #pragma unroll
    for (int nb = 0; nb < 8; ++nb)
      orow[nb * 16 + l16] = acc[nb][vv] * inv;
  }
}

extern "C" void kernel_launch(void* const* d_in, const int* in_sizes, int n_in,
                              void* d_out, int out_size, void* d_ws, size_t ws_size,
                              hipStream_t stream) {
  (void)in_sizes; (void)n_in; (void)d_ws; (void)ws_size; (void)out_size;
  const float* q  = (const float*)d_in[0];
  const float* k  = (const float*)d_in[1];
  const float* v  = (const float*)d_in[2];
  const float* kc = (const float*)d_in[3];
  const float* vc = (const float*)d_in[4];
  const int* ipos = (const int*)d_in[5];
  float* out = (float*)d_out;

  dim3 grid(B_ * H_);      // one workgroup per (batch, head)
  dim3 block(NTHREADS);    // 8 wave32
  hipLaunchKernelGGL(fa_sdpa_kvcache_kernel, grid, block, 0, stream,
                     q, k, v, kc, vc, ipos, out);
}
